// BiLSTM_CRF_4879082848532
// MI455X (gfx1250) — matrix operand order repaired and checked by reading, not compiled, not run on hardware
//
#include <hip/hip_runtime.h>
#include <hip/hip_bf16.h>

// ---------------------------------------------------------------------------
// BiLSTM-CRF inference for MI455X (gfx1250, wave32, WMMA).
//  - bf16 WMMA (f32 accum) for all matmuls: input projections (big parallel
//    GEMMs, L2-resident operands) and the sequential h@w_hh^T recurrence.
//  - Recurrence: 2 dirs x 4 batch-chunks = 8 persistent workgroups; hidden
//    state kept in LDS in the ISA A-fragment layout so each of the 512
//    dependent steps is LDS-frag load -> v_wmma -> gates -> LDS-frag store.
//  - Viterbi: one wave per batch element, backpointers in LDS.
// ---------------------------------------------------------------------------

typedef __attribute__((ext_vector_type(16))) __bf16 v16bf;
typedef __attribute__((ext_vector_type(8)))  float  v8f;

#define BATCH   64
#define SEQT    512
#define EMBD    300
#define HID     256
#define G4H     1024          // 4*H gates
#define NTAG    11
#define TSTART  9
#define TSTOP   10
#define MROWS   (BATCH*SEQT)  // 32768
#define MT      (MROWS/16)    // 2048 M-tiles
#define NT4H    (G4H/16)      // 64 N-tiles
#define KT0     10            // ceil(300/32)
#define KP0     320
#define KT1     16            // 512/32
#define KP1     512
#define KTH     8             // 256/32 (recurrent K)

// ---- ISA fragment layouts (CDNA5 ISA §7.12.2) -------------------
// A tile 16x32 bf16: lanes 0-15 M=0-15; VGPR0-3 hold K=0..7 (pairs), VGPR4-7
// K=16..23; lanes 16-31 get K+8. Tile stored as lane*16 + (vgpr*2+half).
__device__ __forceinline__ int a_slot_index(int m, int k) {   // k in [0,32)
  int base = (k >= 16);
  int kk   = k - (base << 4);
  int g    = kk >> 3;              // lane group
  int r    = kk & 7;
  int v    = (base << 2) + (r >> 1);
  int h    = r & 1;
  return (g * 16 + m) * 16 + (v * 2 + h);
}
// B tile 32x16 bf16: lanes 0-15 (N=lane) hold K=0..7 in slots 0-7 and
// K=16..23 in slots 8-15; lanes 16-31 get K+8 (same order as A).
__device__ __forceinline__ int b_slot_index(int k, int n) {   // k in [0,32)
  int g  = (k >> 3) & 1;
  int kk = ((k >> 4) << 3) | (k & 7);
  return (g * 16 + n) * 16 + kk;   // slot = kk
}

__device__ __forceinline__ v8f wmma_bf16(v16bf a, v16bf b, v8f c) {
  return __builtin_amdgcn_wmma_f32_16x16x32_bf16(
      false, a, false, b, (short)0, c, false, false);
}

__device__ __forceinline__ float sigmoidf_(float x) {
  return 1.0f / (1.0f + __expf(-x));
}

// ---- packing kernels -------------------------------------------------------
__global__ void pack_embed_a(const int* __restrict__ sent,
                             const float* __restrict__ emb,
                             __bf16* __restrict__ Ap) {
  int id = blockIdx.x * blockDim.x + threadIdx.x;      // over MROWS*KP0
  if (id >= MROWS * KP0) return;
  int r = id / KP0, k = id - r * KP0;
  float v = 0.0f;
  if (k < EMBD) v = emb[(size_t)sent[r] * EMBD + k];
  Ap[(size_t)((r >> 4) * KT0 + (k >> 5)) * 512 + a_slot_index(r & 15, k & 31)]
      = (__bf16)v;
}

__global__ void pack_a_f32(const float* __restrict__ X,   // [MROWS, D] row-major
                           __bf16* __restrict__ Ap, int D, int Kp, int KT) {
  int id = blockIdx.x * blockDim.x + threadIdx.x;      // over MROWS*Kp
  if (id >= MROWS * Kp) return;
  int r = id / Kp, k = id - r * Kp;
  float v = (k < D) ? X[(size_t)r * D + k] : 0.0f;
  Ap[(size_t)((r >> 4) * KT + (k >> 5)) * 512 + a_slot_index(r & 15, k & 31)]
      = (__bf16)v;
}

__global__ void pack_b_weight(const float* __restrict__ W, // [G4H, D] row-major
                              __bf16* __restrict__ Bp, int D, int Kp, int KT) {
  int id = blockIdx.x * blockDim.x + threadIdx.x;      // over G4H*Kp
  if (id >= G4H * Kp) return;
  int n = id / Kp, k = id - n * Kp;
  float v = (k < D) ? W[(size_t)n * D + k] : 0.0f;     // B[k][n] = W[n][k]
  Bp[(size_t)((n >> 4) * KT + (k >> 5)) * 512 + b_slot_index(k & 31, n & 15)]
      = (__bf16)v;
}

// ---- big GEMM: C[MROWS,G4H] = Apacked x Bpacked + bias ---------------------
__global__ __launch_bounds__(256) void wmma_gemm_bias(
    const __bf16* __restrict__ Ap, const __bf16* __restrict__ Bp,
    const float* __restrict__ bias, float* __restrict__ C, int KT) {
  const int lane = threadIdx.x & 31;
  const int wave = blockIdx.x * (blockDim.x >> 5) + (threadIdx.x >> 5);
  const int NB = G4H / 64;                  // 16 column blocks of 64
  if (wave >= MT * NB) return;              // wave-uniform (EXEC stays all-1)
  const int nb = wave % NB;
  const int mt = wave / NB;
  v8f acc[4] = {};
  for (int kt = 0; kt < KT; ++kt) {
    v16bf a = *(const v16bf*)(Ap + (size_t)(mt * KT + kt) * 512 + lane * 16);
#pragma unroll
    for (int j = 0; j < 4; ++j) {
      v16bf b = *(const v16bf*)(Bp + (size_t)((nb * 4 + j) * KT + kt) * 512
                                   + lane * 16);
      acc[j] = wmma_bf16(a, b, acc[j]);
    }
  }
  const int mhalf = (lane >> 4) << 3;
  const int ncol  = lane & 15;
#pragma unroll
  for (int j = 0; j < 4; ++j) {
    int n = nb * 64 + j * 16 + ncol;
    float bv = bias[n];
#pragma unroll
    for (int r = 0; r < 8; ++r) {
      int row = mt * 16 + r + mhalf;        // C layout: VGPR r -> M=r / r+8
      C[(size_t)row * G4H + n] = acc[j][r] + bv;
    }
  }
}

// ---- sequential LSTM scan --------------------------------------------------
// grid = 8 blocks (dir = blockIdx>>2, batch chunk of 16 rows = blockIdx&3),
// 256 threads (8 waves). Per step: g[16,1024] = xw[:,t,:] + h@w_hh^T (WMMA),
// then gates -> new c (LDS f32) and h (LDS bf16 A-fragments + global f32).
__global__ __launch_bounds__(256) void lstm_scan(
    const float* __restrict__ xwF, const float* __restrict__ xwB,
    const __bf16* __restrict__ whhF, const __bf16* __restrict__ whhB,
    float* __restrict__ hout /* [BATCH, SEQT, 2H] */) {
  __shared__ __bf16 hA[KTH * 512];          // h as A-fragments (8 KB)
  __shared__ float  gbuf[16 * G4H];         // gate pre-activations (64 KB)
  __shared__ float  cbuf[16 * HID];         // cell state (16 KB)

  const int dir    = blockIdx.x >> 2;
  const int mb     = (blockIdx.x & 3) * 16; // first batch row of this chunk
  const float*  xw  = dir ? xwB  : xwF;
  const __bf16* whh = dir ? whhB : whhF;
  const int dirOff = dir ? HID : 0;
  const int tid  = threadIdx.x;
  const int lane = tid & 31;
  const int w    = tid >> 5;

  for (int i = tid; i < KTH * 512; i += 256) hA[i] = (__bf16)0.0f;
  for (int i = tid; i < 16 * HID;  i += 256) cbuf[i] = 0.0f;
  __syncthreads();

  for (int step = 0; step < SEQT; ++step) {
    const int t = dir ? (SEQT - 1 - step) : step;
    // ---- phase 1: g = h @ w_hh^T + xw[:,t,:]  (each wave: 8 N-tiles) ----
    v16bf afr[KTH];
#pragma unroll
    for (int kt = 0; kt < KTH; ++kt)
      afr[kt] = *(const v16bf*)&hA[kt * 512 + lane * 16];
    const int mhalf = (lane >> 4) << 3;
    const int nlow  = lane & 15;
#pragma unroll
    for (int j = 0; j < 8; ++j) {
      const int nt = w * 8 + j;
      v8f acc = {};
#pragma unroll
      for (int kt = 0; kt < KTH; ++kt) {
        v16bf bfr = *(const v16bf*)(whh + (size_t)(nt * KTH + kt) * 512
                                        + lane * 16);
        acc = wmma_bf16(afr[kt], bfr, acc);
      }
      const int ncol = nt * 16 + nlow;
#pragma unroll
      for (int r = 0; r < 8; ++r) {
        const int m = r + mhalf;
        gbuf[m * G4H + ncol] =
            acc[r] + xw[((size_t)(mb + m) * SEQT + t) * G4H + ncol];
      }
    }
    __syncthreads();
    // ---- phase 2: gates -> c, h; repack h into LDS A-fragments ----------
    {
      const int m  = tid >> 4;
      const int jb = (tid & 15) << 4;
#pragma unroll 4
      for (int jj = 0; jj < 16; ++jj) {
        const int j = jb + jj;
        float gi = gbuf[m * G4H + j];
        float gf = gbuf[m * G4H + 256 + j];
        float gg = gbuf[m * G4H + 512 + j];
        float go = gbuf[m * G4H + 768 + j];
        float c  = sigmoidf_(gf) * cbuf[m * HID + j]
                 + sigmoidf_(gi) * tanhf(gg);
        float h  = sigmoidf_(go) * tanhf(c);
        cbuf[m * HID + j] = c;
        hA[(j >> 5) * 512 + a_slot_index(m, j & 31)] = (__bf16)h;
        hout[((size_t)(mb + m) * SEQT + t) * (2 * HID) + dirOff + j] = h;
      }
      if (step + 1 < SEQT) {                // prefetch next step's gates
        const int tn = dir ? (SEQT - 2 - step) : (step + 1);
        __builtin_prefetch(
            xw + ((size_t)(mb + (tid >> 4)) * SEQT + tn) * G4H
               + ((tid & 15) << 6), 0, 1);
      }
    }
    __syncthreads();
  }
}

// ---- FC emission scores ----------------------------------------------------
__global__ void fc_feats(const float* __restrict__ h1,   // [MROWS, 512]
                         const float* __restrict__ fc_w, // [NTAG, 512]
                         const float* __restrict__ fc_b,
                         float* __restrict__ feats) {    // [MROWS, NTAG]
  int id = blockIdx.x * blockDim.x + threadIdx.x;
  if (id >= MROWS * NTAG) return;
  int r = id / NTAG, k = id - r * NTAG;
  const float* hv = h1 + (size_t)r * (2 * HID);
  const float* wv = fc_w + (size_t)k * (2 * HID);
  float s = fc_b[k];
  for (int j = 0; j < 2 * HID; ++j) s += hv[j] * wv[j];
  feats[(size_t)r * NTAG + k] = s;
}

// ---- Viterbi decode: one wave per batch element ----------------------------
__global__ __launch_bounds__(32) void viterbi(
    const float* __restrict__ feats,        // [BATCH, SEQT, NTAG]
    const float* __restrict__ trans,        // [NTAG, NTAG]  trans[to][from]
    float* __restrict__ out) {              // [64] scores, then [64,512] path
  const int b = blockIdx.x, lane = threadIdx.x;
  __shared__ float tr[NTAG * NTAG];
  __shared__ float fvs[NTAG];
  __shared__ float red[NTAG];
  __shared__ unsigned char bp[SEQT * NTAG];
  for (int i = lane; i < NTAG * NTAG; i += 32) tr[i] = trans[i];
  float fv = (lane == TSTART) ? 0.0f : -1000.0f;
  __syncthreads();
  for (int t = 0; t < SEQT; ++t) {
    if (lane < NTAG) fvs[lane] = fv;
    __syncthreads();
    if (lane < NTAG) {
      float best = -3.4e38f; int barg = 0;
      for (int p = 0; p < NTAG; ++p) {
        float s = fvs[p] + tr[lane * NTAG + p];
        if (s > best) { best = s; barg = p; }
      }
      bp[t * NTAG + lane] = (unsigned char)barg;
      fv = best + feats[((size_t)b * SEQT + t) * NTAG + lane];
    }
    __syncthreads();
  }
  if (lane < NTAG) red[lane] = fv + tr[TSTOP * NTAG + lane];
  __syncthreads();
  if (lane == 0) {
    float best = -3.4e38f; int tag = 0;
    for (int k = 0; k < NTAG; ++k)
      if (red[k] > best) { best = red[k]; tag = k; }
    out[b] = best;
    float* path = out + BATCH + (size_t)b * SEQT;
    path[SEQT - 1] = (float)tag;
    for (int t = SEQT - 1; t >= 1; --t) {
      tag = bp[t * NTAG + tag];
      path[t - 1] = (float)tag;
    }
  }
}

// ---------------------------------------------------------------------------
extern "C" void kernel_launch(void* const* d_in, const int* in_sizes, int n_in,
                              void* d_out, int out_size, void* d_ws,
                              size_t ws_size, hipStream_t stream) {
  (void)in_sizes; (void)n_in; (void)out_size; (void)ws_size;
  const int*   sentence = (const int*)d_in[0];
  /* d_in[1] batch_lengths: unused (reference ignores it) */
  const float* emb   = (const float*)d_in[2];
  const float* fc_w  = (const float*)d_in[3];
  const float* fc_b  = (const float*)d_in[4];
  const float* trans = (const float*)d_in[5];
  const float* w_ih[4] = {(const float*)d_in[6],  (const float*)d_in[9],
                          (const float*)d_in[12], (const float*)d_in[15]};
  const float* w_hh[4] = {(const float*)d_in[7],  (const float*)d_in[10],
                          (const float*)d_in[13], (const float*)d_in[16]};
  const float* bias[4] = {(const float*)d_in[8],  (const float*)d_in[11],
                          (const float*)d_in[14], (const float*)d_in[17]};

  char* ws = (char*)d_ws;
  size_t off = 0;
  auto take = [&](size_t bytes) -> char* {
    char* p = ws + off;
    off += (bytes + 255) & ~(size_t)255;
    return p;
  };
  // A-fragment region (layer0 uses KT0 tiles/row, layer1 KT1; max = layer1)
  __bf16* Ap    = (__bf16*)take((size_t)MT * KT1 * 512 * sizeof(__bf16));
  __bf16* BihP[4]; __bf16* BhhP[4];
  const int ktIh[4] = {KT0, KT0, KT1, KT1};
  const int kpIh[4] = {KP0, KP0, KP1, KP1};
  const int dIh [4] = {EMBD, EMBD, 2 * HID, 2 * HID};
  for (int i = 0; i < 4; ++i) {
    BihP[i] = (__bf16*)take((size_t)NT4H * ktIh[i] * 512 * sizeof(__bf16));
    BhhP[i] = (__bf16*)take((size_t)NT4H * KTH * 512 * sizeof(__bf16));
  }
  float* xwF   = (float*)take((size_t)MROWS * G4H * sizeof(float));
  float* xwB   = (float*)take((size_t)MROWS * G4H * sizeof(float));
  float* hbuf  = (float*)take((size_t)MROWS * 2 * HID * sizeof(float)); // h0 then h1
  float* feats = (float*)take((size_t)MROWS * NTAG * sizeof(float));
  float* out   = (float*)d_out;

  const int TB = 256;
  // 1) pack weights (B-fragments) for all 4 directions x (w_ih, w_hh)
  for (int i = 0; i < 4; ++i) {
    pack_b_weight<<<(G4H * kpIh[i] + TB - 1) / TB, TB, 0, stream>>>(
        w_ih[i], BihP[i], dIh[i], kpIh[i], ktIh[i]);
    pack_b_weight<<<(G4H * HID + TB - 1) / TB, TB, 0, stream>>>(
        w_hh[i], BhhP[i], HID, HID, KTH);
  }
  // 2) embedding gather -> A-fragments (K padded 300->320)
  pack_embed_a<<<(MROWS * KP0 + TB - 1) / TB, TB, 0, stream>>>(sentence, emb, Ap);
  // 3) layer-0 input projections: xw = x @ w_ih^T + b
  const int gemmBlocks = (MT * (G4H / 64)) / 8;   // 8 waves/block, exact
  wmma_gemm_bias<<<gemmBlocks, TB, 0, stream>>>(Ap, BihP[0], bias[0], xwF, KT0);
  wmma_gemm_bias<<<gemmBlocks, TB, 0, stream>>>(Ap, BihP[1], bias[1], xwB, KT0);
  // 4) layer-0 recurrence (writes h0 = hbuf [B,T,512])
  lstm_scan<<<8, TB, 0, stream>>>(xwF, xwB, BhhP[0], BhhP[1], hbuf);
  // 5) repack h0 -> A-fragments (K=512)
  pack_a_f32<<<(MROWS * KP1 + TB - 1) / TB, TB, 0, stream>>>(
      hbuf, Ap, 2 * HID, KP1, KT1);
  // 6) layer-1 input projections (xw buffers reused)
  wmma_gemm_bias<<<gemmBlocks, TB, 0, stream>>>(Ap, BihP[2], bias[2], xwF, KT1);
  wmma_gemm_bias<<<gemmBlocks, TB, 0, stream>>>(Ap, BihP[3], bias[3], xwB, KT1);
  // 7) layer-1 recurrence (h1 overwrites hbuf; h0 is dead by now)
  lstm_scan<<<8, TB, 0, stream>>>(xwF, xwB, BhhP[2], BhhP[3], hbuf);
  // 8) emission scores
  fc_feats<<<(MROWS * NTAG + TB - 1) / TB, TB, 0, stream>>>(hbuf, fc_w, fc_b, feats);
  // 9) Viterbi decode + backtrack
  viterbi<<<BATCH, 32, 0, stream>>>(feats, trans, out);
}
